// fastopic_76544907149767
// MI455X (gfx1250) — compile-verified, matrix operand order checked by reading, not run on hardware
//
#include <hip/hip_runtime.h>

// ---------------------------------------------------------------------------
// FASTopic loss on MI455X (gfx1250): WMMA distance GEMMs + persistent-WGP
// Sinkhorn + fused bf16-WMMA recon/log/dot reduction.
// Dims: N=4096 docs, V=10000 words, K=100 topics (padded KTP=128), E=384.
// ---------------------------------------------------------------------------

#define N_DOCS 4096
#define V_WORDS 10000
#define K_TOP 100
#define KTP 128           // topics padded to 128 (4 x K=32 bf16 WMMA steps)
#define KT2 (KTP / 2)     // packed bf16-pair rows of beta
#define EDIM 384
#define DSR_BLOCKS 20000  // 160000 16x16 tiles / 8 waves per block

typedef float v2f   __attribute__((ext_vector_type(2)));
typedef float v8f   __attribute__((ext_vector_type(8)));
typedef __bf16 v16bf __attribute__((ext_vector_type(16)));
typedef unsigned int v4u __attribute__((ext_vector_type(4)));

union BF16x16 { v16bf v; v4u q[2]; };

__device__ inline float wave_sum(float v) {
  #pragma unroll
  for (int off = 16; off; off >>= 1) v += __shfl_xor(v, off, 32);
  return v;
}

__device__ inline unsigned int f2bf(float x) {
  unsigned int u = __float_as_uint(x);
  return (u + 0x7FFFu + ((u >> 16) & 1u)) >> 16;  // RNE, returned in low 16
}

// ---------------------------------------------------------------------------
// Row squared-norms (rows >= rows_valid get 0).
// ---------------------------------------------------------------------------
__global__ void row_norms_kernel(const float* __restrict__ X, int rows_valid,
                                 int rows_total, float* __restrict__ out) {
  int r = blockIdx.x * blockDim.x + threadIdx.x;
  if (r >= rows_total) return;
  float s = 0.f;
  if (r < rows_valid) {
    const float* p = X + (size_t)r * EDIM;
    for (int e = 0; e < EDIM; ++e) s += p[e] * p[e];
  }
  out[r] = s;
}

// ---------------------------------------------------------------------------
// Softmax over an n-vector (single block, deterministic tree reductions).
// ---------------------------------------------------------------------------
__global__ __launch_bounds__(1024) void softmax_kernel(const float* __restrict__ w,
                                                       int n, float* __restrict__ out) {
  __shared__ float red[1024];
  int tid = threadIdx.x;
  float m = -3.4e38f;
  for (int i = tid; i < n; i += 1024) m = fmaxf(m, w[i]);
  red[tid] = m; __syncthreads();
  for (int s = 512; s; s >>= 1) { if (tid < s) red[tid] = fmaxf(red[tid], red[tid + s]); __syncthreads(); }
  float mx = red[0]; __syncthreads();
  float sum = 0.f;
  for (int i = tid; i < n; i += 1024) sum += __expf(w[i] - mx);
  red[tid] = sum; __syncthreads();
  for (int s = 512; s; s >>= 1) { if (tid < s) red[tid] += red[tid + s]; __syncthreads(); }
  float Z = red[0];
  for (int i = tid; i < n; i += 1024) out[i] = __expf(w[i] - mx) / Z;
}

// ---------------------------------------------------------------------------
// Pairwise sq-euclidean distance + exp(-alpha*M) via V_WMMA_F32_16X16X4_F32.
// One wave per 16x16 output tile, 8 waves/block, E=384 -> 96 WMMAs per tile.
// A-frag (16x4 f32): lane l holds row M=l&15, K pair {4i+2h, 4i+2h+1}, h=l>>4.
// Branchless: OOB rows/cols are clamped to row 0 (their products only land in
// output rows/cols that the epilogue discards), so the inner loop is two
// global_load_b64 + one wmma, no exec-mask branching.
// ---------------------------------------------------------------------------
__global__ __launch_bounds__(256) void dist_wmma_kernel(
    const float* __restrict__ A, const float* __restrict__ B,
    const float* __restrict__ nA, const float* __restrict__ nB,
    int NT, int Avalid, int Bvalid, float alpha,
    float* __restrict__ Mout, float* __restrict__ Kout) {
  int wid = threadIdx.x >> 5, lane = threadIdx.x & 31;
  int ntiles_n = NT >> 4;
  int gtile = blockIdx.x * 8 + wid;
  int tm = gtile / ntiles_n, tn = gtile % ntiles_n;
  int m0 = tm * 16, n0 = tn * 16;
  int h = lane >> 4, q = lane & 15;
  int arow = m0 + q, bcol = n0 + q;
  const float* ap = A + (size_t)(arow < Avalid ? arow : 0) * EDIM + 2 * h;
  const float* bp = B + (size_t)(bcol < Bvalid ? bcol : 0) * EDIM + 2 * h;
  v8f c = {};
  #pragma unroll 4
  for (int k = 0; k < EDIM; k += 4) {
    v2f a = *(const v2f*)(ap + k);   // 8B-aligned pair
    v2f b = *(const v2f*)(bp + k);
    c = __builtin_amdgcn_wmma_f32_16x16x4_f32(false, a, false, b, (short)0, c,
                                              false, false);
  }
  #pragma unroll
  for (int r = 0; r < 8; ++r) {
    int row = m0 + r + 8 * h, col = n0 + q;   // C/D layout: VGPR r -> M=r(+8)
    size_t idx = (size_t)row * NT + col;
    if (row < Avalid && col < Bvalid) {
      float m = nA[row] + nB[col] - 2.f * c[r];
      Mout[idx] = m;
      Kout[idx] = __expf(-alpha * m);
    } else {
      Mout[idx] = 0.f;
      Kout[idx] = 0.f;
    }
  }
}

// ---------------------------------------------------------------------------
// Persistent single-workgroup Sinkhorn, doc-topic (K_dt: 4096 x 128, L2-resident).
// u (4096) and v (128) live in LDS; convergence check every 50 iters as in ref.
// Emits loss_DT and theta = N * u*K*v^T quantized to bf16 (padded cols = 0).
// ---------------------------------------------------------------------------
__global__ __launch_bounds__(1024) void sinkhorn_dt_kernel(
    const float* __restrict__ Kd, const float* __restrict__ Md,
    const float* __restrict__ b_in, unsigned short* __restrict__ theta,
    float* __restrict__ loss_out) {
  __shared__ float s_u[N_DOCS];
  __shared__ float s_v[KTP];
  __shared__ float s_b[KTP];
  __shared__ float s_red[1024];
  __shared__ float s_err[1];
  int tid = threadIdx.x, wid = tid >> 5, lane = tid & 31;
  for (int n = tid; n < N_DOCS; n += 1024) s_u[n] = 1.f / N_DOCS;
  for (int k = tid; k < KTP; k += 1024) {
    s_v[k] = 0.f;
    s_b[k] = (k < K_TOP) ? b_in[k] : 0.f;
  }
  if (tid == 0) s_err[0] = 1.f;
  __syncthreads();
  int cpt = 0;
  while (s_err[0] > 0.005f && cpt < 5000) {
    // v = b / (K^T u + eps): one column per wave
    for (int k = wid; k < K_TOP; k += 32) {
      float s = 0.f;
      for (int n = lane; n < N_DOCS; n += 32) s += Kd[(size_t)n * KTP + k] * s_u[n];
      s = wave_sum(s);
      if (lane == 0) s_v[k] = s_b[k] / (s + 1e-16f);
    }
    __syncthreads();
    // u = a / (K v + eps): rows strided over threads
    for (int n = tid; n < N_DOCS; n += 1024) {
      const float* kr = Kd + (size_t)n * KTP;
      float s = 0.f;
      for (int k = 0; k < K_TOP; ++k) s += kr[k] * s_v[k];
      s_u[n] = (1.f / N_DOCS) / (s + 1e-16f);
    }
    __syncthreads();
    ++cpt;
    if (cpt % 50 == 1) {  // err = sum |v * (K^T u_new) - b|
      for (int k = wid; k < K_TOP; k += 32) {
        float s = 0.f;
        for (int n = lane; n < N_DOCS; n += 32) s += Kd[(size_t)n * KTP + k] * s_u[n];
        s = wave_sum(s);
        if (lane == 0) s_red[k] = fabsf(s_v[k] * s - s_b[k]);
      }
      __syncthreads();
      if (tid == 0) {
        float e = 0.f;
        for (int k = 0; k < K_TOP; ++k) e += s_red[k];
        s_err[0] = e;
      }
      __syncthreads();
    }
  }
  // loss_DT = sum(u*K*v^T * M); theta = N * u*K*v^T  (bf16)
  float acc = 0.f;
  for (int n = tid; n < N_DOCS; n += 1024) {
    float un = s_u[n];
    const float* kr = Kd + (size_t)n * KTP;
    const float* mr = Md + (size_t)n * KTP;
    unsigned short* tr = theta + (size_t)n * KTP;
    for (int k = 0; k < KTP; ++k) {
      float t = un * kr[k] * s_v[k];
      acc += t * mr[k];
      tr[k] = (unsigned short)f2bf(t * (float)N_DOCS);
    }
  }
  s_red[tid] = acc; __syncthreads();
  for (int s = 512; s; s >>= 1) { if (tid < s) s_red[tid] += s_red[tid + s]; __syncthreads(); }
  if (tid == 0) loss_out[0] = s_red[0];
}

// ---------------------------------------------------------------------------
// Persistent single-workgroup Sinkhorn, topic-word (K_tw: 128 x 10000).
// v (10000 floats = 40 KB) fits the 320 KB LDS. Emits loss_TW and beta in a
// WMMA-B-fragment-friendly layout: betaT[col][k/2] = pack(bf16(beta[k][col]),
// bf16(beta[k+1][col])) so the DSR kernel can fetch B-frags with b128 loads.
// ---------------------------------------------------------------------------
__global__ __launch_bounds__(1024) void sinkhorn_tw_kernel(
    const float* __restrict__ Kt, const float* __restrict__ Mt,
    const float* __restrict__ b, unsigned int* __restrict__ betaT,
    float* __restrict__ loss_out) {
  __shared__ float s_v[V_WORDS];
  __shared__ float s_u[KTP];
  __shared__ float s_red[1024];
  __shared__ float s_err[1];
  int tid = threadIdx.x, wid = tid >> 5, lane = tid & 31;
  for (int k = tid; k < KTP; k += 1024) s_u[k] = (k < K_TOP) ? (1.f / K_TOP) : 0.f;
  for (int j = tid; j < V_WORDS; j += 1024) s_v[j] = 0.f;
  if (tid == 0) s_err[0] = 1.f;
  __syncthreads();
  int cpt = 0;
  while (s_err[0] > 0.005f && cpt < 5000) {
    // v_j = b_j / (sum_k K[k,j] u_k + eps): coalesced column sums
    for (int j = tid; j < V_WORDS; j += 1024) {
      float s = 0.f;
      for (int k = 0; k < K_TOP; ++k) s += Kt[(size_t)k * V_WORDS + j] * s_u[k];
      s_v[j] = b[j] / (s + 1e-16f);
    }
    __syncthreads();
    // u_k = a / (sum_j K[k,j] v_j + eps): one row per wave
    for (int k = wid; k < K_TOP; k += 32) {
      const float* kr = Kt + (size_t)k * V_WORDS;
      float s = 0.f;
      for (int j = lane; j < V_WORDS; j += 32) s += kr[j] * s_v[j];
      s = wave_sum(s);
      if (lane == 0) s_u[k] = (1.f / K_TOP) / (s + 1e-16f);
    }
    __syncthreads();
    ++cpt;
    if (cpt % 50 == 1) {
      float e = 0.f;
      for (int j = tid; j < V_WORDS; j += 1024) {
        float s = 0.f;
        for (int k = 0; k < K_TOP; ++k) s += Kt[(size_t)k * V_WORDS + j] * s_u[k];
        e += fabsf(s_v[j] * s - b[j]);
      }
      s_red[tid] = e; __syncthreads();
      for (int s2 = 512; s2; s2 >>= 1) { if (tid < s2) s_red[tid] += s_red[tid + s2]; __syncthreads(); }
      if (tid == 0) s_err[0] = s_red[0];
      __syncthreads();
    }
  }
  float acc = 0.f;
  for (size_t idx2 = tid; idx2 < (size_t)KT2 * V_WORDS; idx2 += 1024) {
    int k2 = (int)(idx2 / V_WORDS), j = (int)(idx2 % V_WORDS);
    int k = 2 * k2;
    size_t i0 = (size_t)k * V_WORDS + j, i1 = i0 + V_WORDS;
    float t0 = s_u[k] * Kt[i0] * s_v[j];
    float t1 = s_u[k + 1] * Kt[i1] * s_v[j];
    acc += t0 * Mt[i0] + t1 * Mt[i1];
    betaT[(size_t)j * KT2 + k2] =
        f2bf(t0 * (float)K_TOP) | (f2bf(t1 * (float)K_TOP) << 16);
  }
  s_red[tid] = acc; __syncthreads();
  for (int s = 512; s; s >>= 1) { if (tid < s) s_red[tid] += s_red[tid + s]; __syncthreads(); }
  if (tid == 0) loss_out[0] = s_red[0];
}

// ---------------------------------------------------------------------------
// Fused recon loss: per 16x16 tile of recon = theta(bf16) @ beta(bf16) via
// 4 x v_wmma_f32_16x16x32_bf16 (K=128). recon never hits memory; the only
// HBM stream is the 164 MB bow read (~7 us at 23.3 TB/s), prefetched ahead of
// the matrix math. A/B fragments are fetched as b128:
//   A (ISA 16-bit A 16x32): lane l: M=l&15, h=l>>4; the 4 bf16-pairs of each
//     half live at ushort offsets 8h+{0,2,4,6} and 16+8h+{0,2,4,6} -> 2x b128.
//   B: betaT[col][k2] packed pairs; per half k2 = kb/2+4h+{0..3} (+8) -> 2x b128.
// Deterministic: per-block partial sums, no FP atomics.
// ---------------------------------------------------------------------------
__global__ __launch_bounds__(256) void dsr_kernel(
    const unsigned short* __restrict__ theta, const unsigned int* __restrict__ betaT,
    const float* __restrict__ bow, float* __restrict__ partials) {
  int wid = threadIdx.x >> 5, lane = threadIdx.x & 31;
  int gtile = blockIdx.x * 8 + wid;                // 160000 tiles total
  int tm = gtile / (V_WORDS / 16), tn = gtile % (V_WORDS / 16);
  int m0 = tm * 16, n0 = tn * 16;
  int h = lane >> 4, q = lane & 15;

  // Prefetch this wave's 16x16 bow tile (global_prefetch_b8).
  __builtin_prefetch(&bow[(size_t)(m0 + q) * V_WORDS + n0 + 8 * h], 0, 1);

  const unsigned short* arow = theta + (size_t)(m0 + q) * KTP + 8 * h;
  const unsigned int* bcol = betaT + (size_t)(n0 + q) * KT2 + 4 * h;
  v8f c = {};
  #pragma unroll
  for (int kb = 0; kb < KTP; kb += 32) {
    BF16x16 a, bm;
    a.q[0] = *(const v4u*)(arow + kb);        // K pairs {8h+0..6}
    a.q[1] = *(const v4u*)(arow + kb + 16);   // K pairs {16+8h+0..6}
    bm.q[0] = *(const v4u*)(bcol + (kb >> 1));
    bm.q[1] = *(const v4u*)(bcol + (kb >> 1) + 8);
    c = __builtin_amdgcn_wmma_f32_16x16x32_bf16(false, a.v, false, bm.v,
                                                (short)0, c, false, false);
  }
  float acc = 0.f;
  #pragma unroll
  for (int r = 0; r < 8; ++r) {
    int row = m0 + r + 8 * h, col = n0 + q;        // C/D layout
    acc += bow[(size_t)row * V_WORDS + col] * __logf(c[r] + 1e-12f);
  }
  acc = wave_sum(acc);
  __shared__ float s[8];
  if (lane == 0) s[wid] = acc;
  __syncthreads();
  if (threadIdx.x == 0) {
    float t = 0.f;
    for (int i = 0; i < 8; ++i) t += s[i];
    partials[blockIdx.x] = t;
  }
}

// ---------------------------------------------------------------------------
// Final combine: loss = -sum(partials)/N + loss_DT + loss_TW.
// ---------------------------------------------------------------------------
__global__ __launch_bounds__(1024) void final_kernel(
    const float* __restrict__ partials, int np, const float* __restrict__ ldt,
    const float* __restrict__ ltw, float* __restrict__ out) {
  __shared__ float red[1024];
  int tid = threadIdx.x;
  float s = 0.f;
  for (int i = tid; i < np; i += 1024) s += partials[i];
  red[tid] = s; __syncthreads();
  for (int t = 512; t; t >>= 1) { if (tid < t) red[tid] += red[tid + t]; __syncthreads(); }
  if (tid == 0) out[0] = -red[0] / (float)N_DOCS + ldt[0] + ltw[0];
}

// ---------------------------------------------------------------------------
extern "C" void kernel_launch(void* const* d_in, const int* in_sizes, int n_in,
                              void* d_out, int out_size, void* d_ws, size_t ws_size,
                              hipStream_t stream) {
  (void)in_sizes; (void)n_in; (void)out_size; (void)ws_size;
  const float* bow    = (const float*)d_in[0];  // [4096,10000]
  const float* doc    = (const float*)d_in[1];  // [4096,384]
  const float* word   = (const float*)d_in[2];  // [10000,384]
  const float* topic  = (const float*)d_in[3];  // [100,384]
  const float* word_w = (const float*)d_in[4];  // [10000,1]
  const float* top_w  = (const float*)d_in[5];  // [100,1]

  // Carve workspace with 64 B alignment (b128 accesses need 16 B).
  char* base = (char*)d_ws;
  size_t off = 0;
  auto take = [&](size_t bytes) -> char* {
    char* p = base + off;
    off += (bytes + 63) & ~(size_t)63;
    return p;
  };
  float* Mdt = (float*)take((size_t)N_DOCS * KTP * 4);
  float* Kdt = (float*)take((size_t)N_DOCS * KTP * 4);
  float* Mtw = (float*)take((size_t)KTP * V_WORDS * 4);
  float* Ktw = (float*)take((size_t)KTP * V_WORDS * 4);
  float* nd  = (float*)take(N_DOCS * 4);
  float* nt  = (float*)take(KTP * 4);
  float* nw  = (float*)take(V_WORDS * 4);
  float* bdt = (float*)take(KTP * 4);
  float* btw = (float*)take(V_WORDS * 4);
  float* ldt = (float*)take(64);
  float* ltw = (float*)take(64);
  float* parts = (float*)take(DSR_BLOCKS * 4);
  unsigned short* theta = (unsigned short*)take((size_t)N_DOCS * KTP * 2);
  unsigned int* betaT   = (unsigned int*)take((size_t)V_WORDS * KT2 * 4);

  row_norms_kernel<<<(N_DOCS + 255) / 256, 256, 0, stream>>>(doc, N_DOCS, N_DOCS, nd);
  row_norms_kernel<<<1, 128, 0, stream>>>(topic, K_TOP, KTP, nt);
  row_norms_kernel<<<(V_WORDS + 255) / 256, 256, 0, stream>>>(word, V_WORDS, V_WORDS, nw);
  softmax_kernel<<<1, 1024, 0, stream>>>(top_w, K_TOP, bdt);
  softmax_kernel<<<1, 1024, 0, stream>>>(word_w, V_WORDS, btw);

  // M_dt / K_dt : 4096x128 -> 2048 tiles -> 256 blocks of 8 waves
  dist_wmma_kernel<<<256, 256, 0, stream>>>(doc, topic, nd, nt,
                                            KTP, N_DOCS, K_TOP, 3.0f, Mdt, Kdt);
  // M_tw / K_tw : 128x10000 -> 5000 tiles -> 625 blocks
  dist_wmma_kernel<<<625, 256, 0, stream>>>(topic, word, nt, nw,
                                            V_WORDS, K_TOP, V_WORDS, 2.0f, Mtw, Ktw);

  sinkhorn_dt_kernel<<<1, 1024, 0, stream>>>(Kdt, Mdt, bdt, theta, ldt);
  sinkhorn_tw_kernel<<<1, 1024, 0, stream>>>(Ktw, Mtw, btw, betaT, ltw);

  dsr_kernel<<<DSR_BLOCKS, 256, 0, stream>>>(theta, betaT, bow, parts);
  final_kernel<<<1, 1024, 0, stream>>>(parts, DSR_BLOCKS, ldt, ltw, (float*)d_out);
}
